// Discriminator_87875030876729
// MI455X (gfx1250) — matrix operand order repaired and checked by reading, not compiled, not run on hardware
//
#include <hip/hip_runtime.h>
#include <hip/hip_bf16.h>

// ---------------------------------------------------------------------------
// Problem constants (from reference)
// ---------------------------------------------------------------------------
#define BB   32
#define SS   512
#define EE   1024
#define HH   8
#define VV   36
#define VP   48            // V padded to 3x16 WMMA tiles
#define MTOT (BB*SS)       // 16384 rows of x / q / k / v
#define FEATN (SS*HH*VV)   // 147456

typedef unsigned short u16;
typedef unsigned int   u32;

// WMMA vector types (CDNA5 gfx1250, wave32)
typedef __attribute__((ext_vector_type(16))) __bf16 v16bf;
typedef __attribute__((ext_vector_type(8)))  float  v8f;

union Frag16 {           // one 16x32 bf16 A or 32x16 bf16 B fragment (8 VGPRs)
    v16bf v;
    u16   u[16];
    u32   w[8];
};

__device__ __forceinline__ v8f wmma_bf16(const Frag16& a, const Frag16& b, v8f c) {
    return __builtin_amdgcn_wmma_f32_16x16x32_bf16(false, a.v, false, b.v,
                                                   (short)0, c, false, false);
}

// round-to-nearest-even float -> bf16 (raw bits)
__device__ __forceinline__ u16 f2bf(float f) {
    u32 u = __float_as_uint(f);
    u = (u + 0x7FFFu + ((u >> 16) & 1u)) >> 16;
    return (u16)u;
}
__device__ __forceinline__ u32 f2bf2(float lo, float hi) {
    return (u32)f2bf(lo) | ((u32)f2bf(hi) << 16);
}

// A-fragment K offset within a 32-deep K block (contiguous: k = 16*half + 2*vg, matches B)
__device__ __forceinline__ int akoff(int vg, int half) {
    return 2 * vg + 16 * half;
}

// CDNA5 async copy: 16 bytes/lane global -> LDS, tracked by ASYNCcnt.
// LDS address = low 32 bits of the flat shared pointer (aperture mapping).
__device__ __forceinline__ void async_copy_b128(const u16* gsrc, void* ldst) {
    u32 loff = (u32)(size_t)ldst;
    unsigned long long ga = (unsigned long long)(size_t)gsrc;
    asm volatile("global_load_async_to_lds_b128 %0, %1, off"
                 :: "v"(loff), "v"(ga) : "memory");
}

// ---------------------------------------------------------------------------
// 1) Embedding gather -> bf16 x [MTOT][EE]
// ---------------------------------------------------------------------------
__global__ __launch_bounds__(256) void gather_kernel(const int* __restrict__ tokens,
                                                     const float* __restrict__ emb,
                                                     u16* __restrict__ x) {
    int bs  = blockIdx.x;
    int tok = tokens[bs];
    const float* row = emb + (size_t)tok * EE;
    u16*         xr  = x   + (size_t)bs  * EE;
    int e = threadIdx.x * 4;
    float4 f = *(const float4*)(row + e);
    uint2 pk; pk.x = f2bf2(f.x, f.y); pk.y = f2bf2(f.z, f.w);
    *(uint2*)(xr + e) = pk;
}

// ---------------------------------------------------------------------------
// 2a) Wq/Wk [H][K][N] fp32 -> bf16 transposed WT [H][N][K] (LDS-tiled, coalesced)
// ---------------------------------------------------------------------------
__global__ __launch_bounds__(256) void cvt_wT_kernel(const float* __restrict__ W,
                                                     u16* __restrict__ WT) {
    __shared__ u16 tile[32][34];
    int h  = blockIdx.z;
    int nb = blockIdx.x * 32, kb = blockIdx.y * 32;
    int tx = threadIdx.x & 31, ty = threadIdx.x >> 5;   // 32 x 8
    #pragma unroll
    for (int i = 0; i < 4; ++i) {
        int k = kb + ty + i * 8;
        tile[ty + i * 8][tx] = f2bf(W[((size_t)h * EE + k) * EE + nb + tx]);
    }
    __syncthreads();
    #pragma unroll
    for (int i = 0; i < 4; ++i) {
        int n = nb + ty + i * 8;
        WT[((size_t)h * EE + n) * EE + kb + tx] = tile[tx][ty + i * 8];
    }
}

// 2b) Wv [H][EE][VV] -> bf16 transposed + padded Wvt [H][VP][EE]
__global__ __launch_bounds__(256) void cvt_wvt_kernel(const float* __restrict__ Wv,
                                                      u16* __restrict__ wvt) {
    int blk = blockIdx.x;
    int h = blk / VP, vv = blk % VP;
    int e0 = threadIdx.x * 4;
    u16* o = wvt + ((size_t)h * VP + vv) * EE + e0;
    #pragma unroll
    for (int j = 0; j < 4; ++j) {
        float val = (vv < VV) ? Wv[((size_t)h * EE + (e0 + j)) * VV + vv] : 0.0f;
        o[j] = f2bf(val);
    }
}

// ---------------------------------------------------------------------------
// 3) Projection GEMM: Out[h] = relu(x @ W[h] + b[h]), bf16 out [H][MTOT][N]
//    128x128 block tile, 8 waves (2M x 4N) of 64x32, K-step 32.
//    Double-buffered async global->LDS staging (ASYNCcnt pipeline).
// ---------------------------------------------------------------------------
__global__ __launch_bounds__(256) void proj_kernel(const u16* __restrict__ X,
                                                   const u16* __restrict__ WT,
                                                   const float* __restrict__ bias,
                                                   u16* __restrict__ Out) {
    constexpr int N = EE, K = EE, STAGES = K / 32;
    __shared__ __align__(16) u16 Asmem[2][128 * 32];   // [m][k]
    __shared__ __align__(16) u16 Bsmem[2][128 * 32];   // [n][k]

    int h = blockIdx.z;
    const u16*   Wh = WT   + (size_t)h * (size_t)K * N;   // transposed [n][k]
    const float* bh = bias + (size_t)h * N;
    u16*         Oh = Out  + (size_t)h * (size_t)MTOT * N;

    int nBase = blockIdx.x * 128, mBase = blockIdx.y * 128;
    int t = threadIdx.x;
    int wave = t >> 5, lane = t & 31;
    int wm = wave & 1, wn = wave >> 1;
    int m = lane & 15, half = lane >> 4;

    int idx0 = t * 16, idx1 = t * 16 + 8;   // staging element indices (b128 each)

    auto issue = [&](int s, int buf) {       // 4 async b128 per thread per stage
        int k0 = s * 32;
        async_copy_b128(X  + (size_t)(mBase + (idx0 >> 5)) * K + k0 + (idx0 & 31),
                        &Asmem[buf][idx0]);
        async_copy_b128(X  + (size_t)(mBase + (idx1 >> 5)) * K + k0 + (idx1 & 31),
                        &Asmem[buf][idx1]);
        async_copy_b128(Wh + (size_t)(nBase + (idx0 >> 5)) * K + k0 + (idx0 & 31),
                        &Bsmem[buf][idx0]);
        async_copy_b128(Wh + (size_t)(nBase + (idx1 >> 5)) * K + k0 + (idx1 & 31),
                        &Bsmem[buf][idx1]);
    };

    v8f acc[4][2] = {};
    issue(0, 0);
    for (int s = 0; s < STAGES; ++s) {
        int buf = s & 1;
        if (s + 1 < STAGES) {
            issue(s + 1, buf ^ 1);
            asm volatile("s_wait_asynccnt 0x4" ::: "memory"); // stage s resident
        } else {
            asm volatile("s_wait_asynccnt 0x0" ::: "memory");
        }
        __syncthreads();

        Frag16 a[4];
        #pragma unroll
        for (int fm = 0; fm < 4; ++fm)
            #pragma unroll
            for (int vg = 0; vg < 8; ++vg)
                a[fm].w[vg] = *(const u32*)&Asmem[buf][(64 * wm + 16 * fm + m) * 32
                                                       + akoff(vg, half)];
        Frag16 bf[2];
        #pragma unroll
        for (int fn = 0; fn < 2; ++fn)
            #pragma unroll
            for (int vg = 0; vg < 8; ++vg)
                bf[fn].w[vg] = *(const u32*)&Bsmem[buf][(32 * wn + 16 * fn + m) * 32
                                                        + 16 * half + 2 * vg];
        #pragma unroll
        for (int fm = 0; fm < 4; ++fm)
            #pragma unroll
            for (int fn = 0; fn < 2; ++fn)
                acc[fm][fn] = wmma_bf16(a[fm], bf[fn], acc[fm][fn]);
        __syncthreads();
    }

    // bias + relu + bf16 store.  C layout: row = r + 8*half, col = lane&15.
    #pragma unroll
    for (int fn = 0; fn < 2; ++fn) {
        int col = nBase + 32 * wn + 16 * fn + m;
        float bv = bh[col];
        #pragma unroll
        for (int fm = 0; fm < 4; ++fm) {
            #pragma unroll
            for (int r = 0; r < 8; ++r) {
                int row = mBase + 64 * wm + 16 * fm + r + 8 * half;
                float val = fmaxf(acc[fm][fn][r] + bv, 0.0f);
                Oh[(size_t)row * N + col] = f2bf(val);
            }
        }
    }
}

// ---------------------------------------------------------------------------
// 4) V projection: vT[h][vv][row] = relu(x @ Wv[h] + bv)  (bf16, transposed)
// ---------------------------------------------------------------------------
__global__ __launch_bounds__(256) void vproj_kernel(const u16* __restrict__ X,
                                                    const u16* __restrict__ Wvt,
                                                    const float* __restrict__ bv,
                                                    u16* __restrict__ vT) {
    int h = blockIdx.y;
    int wave = threadIdx.x >> 5, lane = threadIdx.x & 31;
    int m = lane & 15, half = lane >> 4;
    int rowBase = blockIdx.x * 128 + wave * 16;
    const u16* Wh = Wvt + (size_t)h * VP * EE;

    v8f acc[3] = {};
    for (int k0 = 0; k0 < EE; k0 += 32) {
        Frag16 a;
        const u16* xr = X + (size_t)(rowBase + m) * EE + k0;
        #pragma unroll
        for (int vg = 0; vg < 8; ++vg) a.w[vg] = *(const u32*)(xr + akoff(vg, half));
        #pragma unroll
        for (int fv = 0; fv < 3; ++fv) {
            Frag16 bf;
            const u16* wr = Wh + (size_t)(fv * 16 + m) * EE + k0 + 16 * half;
            #pragma unroll
            for (int vg = 0; vg < 8; ++vg) bf.w[vg] = *(const u32*)(wr + 2 * vg);
            acc[fv] = wmma_bf16(a, bf, acc[fv]);
        }
    }
    #pragma unroll
    for (int fv = 0; fv < 3; ++fv) {
        int vv = fv * 16 + m;
        float bb = (vv < VV) ? bv[h * VV + vv] : 0.0f;
        #pragma unroll
        for (int r = 0; r < 8; ++r) {
            int row = rowBase + r + 8 * half;
            float val = fmaxf(acc[fv][r] + bb, 0.0f);
            vT[((size_t)h * VP + vv) * MTOT + row] = f2bf(val);
        }
    }
}

// ---------------------------------------------------------------------------
// 5) Flash attention per (h, b, 128-row tile): wave owns 16 query rows.
//    K blocks staged once per workgroup via double-buffered async LDS copies
//    (removes the 8x per-wave redundant K traffic), online softmax, P@V WMMA.
// ---------------------------------------------------------------------------
__global__ __launch_bounds__(256) void attn_kernel(const u16* __restrict__ qb,
                                                   const u16* __restrict__ kb,
                                                   const u16* __restrict__ vT,
                                                   float* __restrict__ feat) {
    constexpr int KSTAGES = EE / 32;
    __shared__ __align__(16) u16 Ksmem[2][64 * 32];   // [t][k] K-block tile
    __shared__ __align__(16) u16 Plds[8][16][64];     // per-wave P scratch (bf16)

    int h = blockIdx.z, b = blockIdx.y;
    int t = threadIdx.x;
    int wave = t >> 5, lane = t & 31;
    int m = lane & 15, half = lane >> 4;
    int sBase = blockIdx.x * 128 + wave * 16;

    const u16* q  = qb + ((size_t)h * MTOT + (size_t)b * SS) * EE;
    const u16* k  = kb + ((size_t)h * MTOT + (size_t)b * SS) * EE;
    const u16* vh = vT + (size_t)h * VP * MTOT + (size_t)b * SS;

    int sidx = t * 8;                       // one b128 per thread per stage
    auto issueK = [&](int tb, int s, int buf) {
        int k0 = s * 32;
        async_copy_b128(k + (size_t)(tb + (sidx >> 5)) * EE + k0 + (sidx & 31),
                        &Ksmem[buf][sidx]);
    };

    float rm[8], rs[8];
    #pragma unroll
    for (int r = 0; r < 8; ++r) { rm[r] = -1e30f; rs[r] = 0.0f; }
    v8f O[3] = {};

    for (int tb = 0; tb < SS; tb += 64) {
        v8f Sa[4] = {};
        issueK(tb, 0, 0);
        for (int s = 0; s < KSTAGES; ++s) {
            int buf = s & 1;
            if (s + 1 < KSTAGES) {
                issueK(tb, s + 1, buf ^ 1);
                asm volatile("s_wait_asynccnt 0x1" ::: "memory");
            } else {
                asm volatile("s_wait_asynccnt 0x0" ::: "memory");
            }
            __syncthreads();

            int k0 = s * 32;
            Frag16 a;
            const u16* qrow = q + (size_t)(sBase + m) * EE + k0;
            #pragma unroll
            for (int vg = 0; vg < 8; ++vg) a.w[vg] = *(const u32*)(qrow + akoff(vg, half));
            #pragma unroll
            for (int fn = 0; fn < 4; ++fn) {
                Frag16 bf;   // B[k][n] = K[t = tb+fn*16+n][e] from LDS tile
                #pragma unroll
                for (int vg = 0; vg < 8; ++vg)
                    bf.w[vg] = *(const u32*)&Ksmem[buf][(fn * 16 + m) * 32
                                                        + 16 * half + 2 * vg];
                Sa[fn] = wmma_bf16(a, bf, Sa[fn]);
            }
            __syncthreads();
        }
        // scale by 1/sqrt(E) = 1/32
        #pragma unroll
        for (int fn = 0; fn < 4; ++fn)
            #pragma unroll
            for (int r = 0; r < 8; ++r) Sa[fn][r] *= 0.03125f;

        // online softmax update (row = r + 8*half, cols across lanes of the half)
        #pragma unroll
        for (int r = 0; r < 8; ++r) {
            float x = fmaxf(fmaxf(Sa[0][r], Sa[1][r]), fmaxf(Sa[2][r], Sa[3][r]));
            x = fmaxf(x, __shfl_xor(x, 1, 32));
            x = fmaxf(x, __shfl_xor(x, 2, 32));
            x = fmaxf(x, __shfl_xor(x, 4, 32));
            x = fmaxf(x, __shfl_xor(x, 8, 32));
            float nm = fmaxf(rm[r], x);
            float sc = __expf(rm[r] - nm);
            rm[r] = nm;
            int row = r + 8 * half;
            float sum = 0.0f;
            #pragma unroll
            for (int fn = 0; fn < 4; ++fn) {
                float p = __expf(Sa[fn][r] - nm);
                sum += p;
                Plds[wave][row][fn * 16 + m] = f2bf(p);
            }
            sum += __shfl_xor(sum, 1, 32);
            sum += __shfl_xor(sum, 2, 32);
            sum += __shfl_xor(sum, 4, 32);
            sum += __shfl_xor(sum, 8, 32);
            rs[r] = rs[r] * sc + sum;
            O[0][r] *= sc; O[1][r] *= sc; O[2][r] *= sc;
        }
        // wave-local LDS RAW fence before reading P as an A fragment
        asm volatile("s_wait_dscnt 0x0" ::: "memory");

        // O += P[16x64] @ V[64x48]
        #pragma unroll
        for (int kbb = 0; kbb < 64; kbb += 32) {
            Frag16 a;
            #pragma unroll
            for (int vg = 0; vg < 8; ++vg)
                a.w[vg] = *(const u32*)&Plds[wave][m][kbb + akoff(vg, half)];
            #pragma unroll
            for (int fv = 0; fv < 3; ++fv) {
                Frag16 bf;   // B[k][n] = v[t][vv] via transposed vT (pairs contiguous)
                const u16* vp = vh + (size_t)(fv * 16 + m) * MTOT + tb + kbb + 16 * half;
                #pragma unroll
                for (int vg = 0; vg < 8; ++vg) bf.w[vg] = *(const u32*)(vp + 2 * vg);
                O[fv] = wmma_bf16(a, bf, O[fv]);
            }
        }
    }

    // normalize and scatter to feat[b][s][h][vv]
    float* fb = feat + (size_t)b * FEATN;
    #pragma unroll
    for (int fv = 0; fv < 3; ++fv) {
        int vv = fv * 16 + m;
        if (vv < VV) {
            #pragma unroll
            for (int r = 0; r < 8; ++r) {
                int s = sBase + r + 8 * half;
                fb[((size_t)s * HH + h) * VV + vv] = O[fv][r] / rs[r];
            }
        }
    }
}

// ---------------------------------------------------------------------------
// 6) Final FC (147456 -> 2) + sigmoid + log_softmax.  d_out = [pred(64), score(64)]
// ---------------------------------------------------------------------------
__global__ __launch_bounds__(256) void fc_kernel(const float* __restrict__ feat,
                                                 const float* __restrict__ fcw,
                                                 const float* __restrict__ fcb,
                                                 float* __restrict__ out) {
    __shared__ float s0[256], s1[256];
    int b = blockIdx.x, tid = threadIdx.x;
    const float* f = feat + (size_t)b * FEATN;
    float a0 = 0.0f, a1 = 0.0f;
    for (int j = tid; j < FEATN; j += 256) {
        float x = f[j];
        a0 += x * fcw[j];
        a1 += x * fcw[FEATN + j];
    }
    s0[tid] = a0; s1[tid] = a1;
    __syncthreads();
    for (int st = 128; st > 0; st >>= 1) {
        if (tid < st) { s0[tid] += s0[tid + st]; s1[tid] += s1[tid + st]; }
        __syncthreads();
    }
    if (tid == 0) {
        float l0 = s0[0] + fcb[0], l1 = s1[0] + fcb[1];
        float sc0 = 1.0f / (1.0f + __expf(-l0));
        float sc1 = 1.0f / (1.0f + __expf(-l1));
        float mx  = fmaxf(sc0, sc1);
        float lse = mx + __logf(__expf(sc0 - mx) + __expf(sc1 - mx));
        out[b * 2 + 0] = sc0 - lse;
        out[b * 2 + 1] = sc1 - lse;
        out[64 + b * 2 + 0] = sc0;
        out[64 + b * 2 + 1] = sc1;
    }
}

// ---------------------------------------------------------------------------
// Workspace layout (bytes, all 256-aligned); ~607 MB total
// ---------------------------------------------------------------------------
static constexpr size_t OFF_X    = 0;                                   // 33.5 MB
static constexpr size_t OFF_WQ   = OFF_X    + (size_t)MTOT * EE * 2;    // 16.8 MB (transposed)
static constexpr size_t OFF_WK   = OFF_WQ   + (size_t)HH * EE * EE * 2; // 16.8 MB (transposed)
static constexpr size_t OFF_WVT  = OFF_WK   + (size_t)HH * EE * EE * 2; // 0.8 MB
static constexpr size_t OFF_Q    = OFF_WVT  + (size_t)HH * VP * EE * 2; // 268 MB
static constexpr size_t OFF_K    = OFF_Q    + (size_t)HH * MTOT * EE * 2;
static constexpr size_t OFF_VT   = OFF_K    + (size_t)HH * MTOT * EE * 2;
static constexpr size_t OFF_FEAT = OFF_VT   + (size_t)HH * VP * MTOT * 2;

extern "C" void kernel_launch(void* const* d_in, const int* in_sizes, int n_in,
                              void* d_out, int out_size, void* d_ws, size_t ws_size,
                              hipStream_t stream) {
    const int*   tokens = (const int*)  d_in[0];
    const float* emb    = (const float*)d_in[1];
    const float* Wq     = (const float*)d_in[2];
    const float* bq     = (const float*)d_in[3];
    const float* Wk     = (const float*)d_in[4];
    const float* bk     = (const float*)d_in[5];
    const float* Wv     = (const float*)d_in[6];
    const float* bv     = (const float*)d_in[7];
    const float* fcw    = (const float*)d_in[8];
    const float* fcb    = (const float*)d_in[9];
    float* out = (float*)d_out;

    char* ws   = (char*)d_ws;
    u16*   x    = (u16*)  (ws + OFF_X);
    u16*   wqT  = (u16*)  (ws + OFF_WQ);
    u16*   wkT  = (u16*)  (ws + OFF_WK);
    u16*   wvt  = (u16*)  (ws + OFF_WVT);
    u16*   qbuf = (u16*)  (ws + OFF_Q);
    u16*   kbuf = (u16*)  (ws + OFF_K);
    u16*   vtb  = (u16*)  (ws + OFF_VT);
    float* feat = (float*)(ws + OFF_FEAT);

    // 1) embedding gather -> bf16
    gather_kernel<<<MTOT, 256, 0, stream>>>(tokens, emb, x);
    // 2) weight converts (Wq/Wk transposed for async B staging)
    cvt_wT_kernel<<<dim3(EE / 32, EE / 32, HH), 256, 0, stream>>>(Wq, wqT);
    cvt_wT_kernel<<<dim3(EE / 32, EE / 32, HH), 256, 0, stream>>>(Wk, wkT);
    cvt_wvt_kernel<<<HH * VP, 256, 0, stream>>>(Wv, wvt);
    // 3) Q and K projections (relu(xW+b)), bf16 out
    proj_kernel<<<dim3(EE / 128, MTOT / 128, HH), 256, 0, stream>>>(x, wqT, bq, qbuf);
    proj_kernel<<<dim3(EE / 128, MTOT / 128, HH), 256, 0, stream>>>(x, wkT, bk, kbuf);
    // 4) V projection (transposed bf16 out)
    vproj_kernel<<<dim3(MTOT / 128, HH), 256, 0, stream>>>(x, wvt, bv, vtb);
    // 5) attention
    attn_kernel<<<dim3(SS / 128, BB, HH), 256, 0, stream>>>(qbuf, kbuf, vtb, feat);
    // 6) final FC + activations
    fc_kernel<<<BB, 256, 0, stream>>>(feat, fcw, fcb, out);
}
